// DiffusionPriorNetwork_16243566314103
// MI455X (gfx1250) — matrix operand order repaired and checked by reading, not compiled, not run on hardware
//
#include <hip/hip_runtime.h>

typedef unsigned int  u32;
typedef unsigned short u16;
typedef __attribute__((ext_vector_type(16))) __bf16 v16bf;
typedef __attribute__((ext_vector_type(8)))  float  v8f;
typedef __attribute__((ext_vector_type(8)))  u32    v8u;

#define DIMM   768
#define HEADS  12
#define DH     64
#define NTOK   81
#define NPAD   96
#define BATCH  256
#define ROWS   (BATCH * NTOK)   /* 20736 */
#define FFI    3072
#define NEGF   (-3.4e38f)

#define AS3 __attribute__((address_space(3)))
#define AS1 __attribute__((address_space(1)))

/* ---- optional CDNA5 async global->LDS path (guarded; falls back to ds copy) ---- */
#if defined(__has_builtin)
#  if __has_builtin(__builtin_amdgcn_global_load_async_to_lds_b32)
#    define HAVE_ASYNC_LDS 1
#  endif
#endif

#ifdef HAVE_ASYNC_LDS
static __device__ __forceinline__ void async_cp_b32(u32* dst_lds, const u32* src_glob) {
  __builtin_amdgcn_global_load_async_to_lds_b32((AS1 int*)src_glob,
                                                (AS3 int*)dst_lds, 0, 0);
}
#  if __has_builtin(__builtin_amdgcn_s_wait_asynccnt)
#    define ASYNC_WAIT() __builtin_amdgcn_s_wait_asynccnt(0)
#  else
#    define ASYNC_WAIT() asm volatile("s_wait_asynccnt 0" ::: "memory")
#  endif
#endif

/* ---------------- helpers ---------------- */

static __device__ __forceinline__ u16 f2bf(float f) {
  u32 u = __builtin_bit_cast(u32, f);
  u32 r = (u + 0x7FFFu + ((u >> 16) & 1u)) >> 16;   /* round-to-nearest-even */
  return (u16)r;
}
static __device__ __forceinline__ u32 pk2(float a, float b) {
  return (u32)f2bf(a) | ((u32)f2bf(b) << 16);
}
static __device__ __forceinline__ v8f zero8() {
  v8f z;
#pragma unroll
  for (int i = 0; i < 8; ++i) z[i] = 0.f;
  return z;
}
static __device__ __forceinline__ v8f wmma_acc(v16bf a, v16bf b, v8f c) {
  return __builtin_amdgcn_wmma_f32_16x16x32_bf16(false, a, false, b, (short)0, c,
                                                 false, false);
}

/* A-fragment (16x32, MxK): lane m=l&15, k = 16*(l>>4) + 2*j.
   LDS layout s[m][k], row stride ST halves (even). Pairs (k,k+1) adjacent -> u32. */
static __device__ __forceinline__ v16bf ldsA(const u16* s, int ST, int m0, int kk, int lane) {
  const u32* su = (const u32*)s;
  int m = m0 + (lane & 15);
  int base = (m * ST + kk + ((lane >> 4) << 4)) >> 1;
  v8u r;
#pragma unroll
  for (int j = 0; j < 8; ++j) r[j] = su[base + j];
  return __builtin_bit_cast(v16bf, r);
}
/* B-fragment (32x16, KxN) from K-pair-packed LDS: s[kp][n] u32 = (W[2kp][n], W[2kp+1][n]).
   lane n=l&15, k = 16*(l>>4) + 2j  ->  kp = 8*(l>>4) + j. Single u32 load per VGPR. */
static __device__ __forceinline__ v16bf ldsBP(const u32* s, int STu, int n0, int kk, int lane) {
  int n = n0 + (lane & 15);
  int kp = (kk >> 1) + ((lane >> 4) << 3);
  v8u r;
#pragma unroll
  for (int j = 0; j < 8; ++j) r[j] = s[(kp + j) * STu + n];
  return __builtin_bit_cast(v16bf, r);
}
/* B-fragment when LDS holds B^T, i.e. s[n][k] with pairs adjacent (q@k^T path). */
static __device__ __forceinline__ v16bf ldsBT(const u16* s, int ST, int n0, int kk, int lane) {
  const u32* su = (const u32*)s;
  int n = n0 + (lane & 15);
  int base = (n * ST + kk + ((lane >> 4) << 4)) >> 1;
  v8u r;
#pragma unroll
  for (int j = 0; j < 8; ++j) r[j] = su[base + j];
  return __builtin_bit_cast(v16bf, r);
}

static __device__ float blk_sum(float v, float* sbuf) {
#pragma unroll
  for (int o = 16; o > 0; o >>= 1) v += __shfl_down(v, o, 32);
  int lane = threadIdx.x & 31, wv = threadIdx.x >> 5;
  __syncthreads();
  if (lane == 0) sbuf[wv] = v;
  __syncthreads();
  if (threadIdx.x == 0) {
    float r = 0.f;
    for (int i = 0; i < 8; ++i) r += sbuf[i];
    sbuf[0] = r;
  }
  __syncthreads();
  return sbuf[0];
}

/* ---------------- token embedding assembly ---------------- */

__global__ __launch_bounds__(256) void k_embed(
    const float* __restrict__ text_enc, const float* __restrict__ text_emb,
    const float* __restrict__ time_emb, const int* __restrict__ tsteps,
    const float* __restrict__ img_emb, const float* __restrict__ lq,
    float* __restrict__ x) {
  int p = blockIdx.x, b = blockIdx.y;
  float* xr = x + ((size_t)b * NTOK + p) * DIMM;
#pragma unroll
  for (int i = 0; i < 3; ++i) {
    int c = threadIdx.x + i * 256;
    float v;
    if (p < 77)       v = text_enc[((size_t)b * 77 + p) * DIMM + c];
    else if (p == 77) v = text_emb[(size_t)b * DIMM + c];
    else if (p == 78) v = time_emb[(size_t)tsteps[b] * DIMM + c];
    else if (p == 79) v = img_emb[(size_t)b * DIMM + c];
    else              v = lq[c];
    xr[c] = v;
  }
}

/* ---------------- relative position bias table ---------------- */

__global__ __launch_bounds__(96) void k_bias(const float* __restrict__ table,
                                             float* __restrict__ bias) {
  int i = blockIdx.x;      /* 0..80 */
  int j = threadIdx.x;     /* 0..95 */
  if (j >= 82) return;
  int nn = i - j; if (nn < 0) nn = 0;
  int bucket;
  if (nn < 16) bucket = nn;
  else {
    int lg = 16 + (int)(logf((float)nn / 16.f) * (16.f / logf(8.f)));
    bucket = lg < 31 ? lg : 31;
  }
#pragma unroll
  for (int h = 0; h < HEADS; ++h)
    bias[((size_t)h * NTOK + i) * 82 + j] = table[bucket * HEADS + h];
}

/* ---------------- layernorm (optionally residual-add) ---------------- */

__global__ __launch_bounds__(256) void k_ln(const float* __restrict__ in, long in_rs,
                                            const float* __restrict__ g,
                                            float* __restrict__ out, long out_rs,
                                            int addTo) {
  __shared__ float sbuf[8];
  long row = blockIdx.x;
  const float* ir = in + row * in_rs;
  int c0 = threadIdx.x, c1 = c0 + 256, c2 = c0 + 512;
  float v0 = ir[c0], v1 = ir[c1], v2 = ir[c2];
  float mu = blk_sum(v0 + v1 + v2, sbuf) * (1.f / 768.f);
  float d0 = v0 - mu, d1 = v1 - mu, d2 = v2 - mu;
  float var = blk_sum(d0 * d0 + d1 * d1 + d2 * d2, sbuf) * (1.f / 768.f);
  float r = rsqrtf(var + 1e-5f);
  float* orow = out + row * out_rs;
  float y0 = d0 * r * g[c0], y1 = d1 * r * g[c1], y2 = d2 * r * g[c2];
  if (addTo) { orow[c0] += y0; orow[c1] += y1; orow[c2] += y2; }
  else       { orow[c0]  = y0; orow[c1]  = y1; orow[c2]  = y2; }
}

/* ---------------- main bf16 WMMA GEMM: C = A(MxK) @ W(KxN) ----------------
   ABF16: A stored as bf16 (else f32, converted while staging to LDS)
   EPI:   0 -> C[idx] = acc ; 1 -> C[idx] += acc (residual)                 */

template <int ABF16, int EPI>
__global__ __launch_bounds__(128) void k_gemm(const void* __restrict__ Ap,
                                              const float* __restrict__ W,
                                              float* __restrict__ C,
                                              int M, int K, int N) {
  __shared__ u16 As[64 * 66];
  __shared__ u32 Wp[32 * 65];            /* K-pair packed W tile */
  const int tid = threadIdx.x, lane = tid & 31, wv = tid >> 5;
  const int m0 = blockIdx.y * 64, n0 = blockIdx.x * 64;
  const int mh = (wv & 1) * 32, nh = (wv >> 1) * 32;
  v8f c[2][2];
#pragma unroll
  for (int a = 0; a < 2; ++a)
#pragma unroll
    for (int b2 = 0; b2 < 2; ++b2) c[a][b2] = zero8();

  const int r = tid >> 1, ch = (tid & 1) * 32;     /* A staging */
  const int kp = tid >> 2, nb = (tid & 3) * 16;    /* W staging */
  for (int k0 = 0; k0 < K; k0 += 64) {
    /* stage A tile (64 rows x 64 k) */
    u32* da = (u32*)As + ((r * 66 + ch) >> 1);
    if (ABF16) {
      const u32* src = (const u32*)((const u16*)Ap + (size_t)(m0 + r) * K + k0 + ch);
#pragma unroll
      for (int i = 0; i < 16; ++i) da[i] = src[i];
      if (k0 + 64 < K) __builtin_prefetch((const char*)src + 128, 0, 1);
    } else {
      const float* src = (const float*)Ap + (size_t)(m0 + r) * K + k0 + ch;
#pragma unroll
      for (int i = 0; i < 8; ++i) {
        float4 v4 = ((const float4*)src)[i];
        da[i * 2]     = pk2(v4.x, v4.y);
        da[i * 2 + 1] = pk2(v4.z, v4.w);
      }
      if (k0 + 64 < K) __builtin_prefetch((const char*)src + 256, 0, 1);
    }
    /* stage W tile K-pair packed: Wp[kp][n] = (W[k0+2kp][n], W[k0+2kp+1][n]) */
    {
      const float* r0 = W + (size_t)(k0 + 2 * kp) * N + n0 + nb;
      const float* r1 = r0 + N;
      u32* dw = Wp + kp * 65 + nb;
#pragma unroll
      for (int i = 0; i < 4; ++i) {
        float4 a4 = ((const float4*)r0)[i];
        float4 b4 = ((const float4*)r1)[i];
        dw[i * 4 + 0] = pk2(a4.x, b4.x);
        dw[i * 4 + 1] = pk2(a4.y, b4.y);
        dw[i * 4 + 2] = pk2(a4.z, b4.z);
        dw[i * 4 + 3] = pk2(a4.w, b4.w);
      }
      if (k0 + 64 < K) __builtin_prefetch((const char*)(r0 + (size_t)64 * N), 0, 1);
    }
    __syncthreads();
#pragma unroll
    for (int kk = 0; kk < 64; kk += 32) {
      v16bf a0 = ldsA(As, 66, mh, kk, lane);
      v16bf a1 = ldsA(As, 66, mh + 16, kk, lane);
      v16bf b0 = ldsBP(Wp, 65, nh, kk, lane);
      v16bf b1 = ldsBP(Wp, 65, nh + 16, kk, lane);
      c[0][0] = wmma_acc(a0, b0, c[0][0]);
      c[0][1] = wmma_acc(a0, b1, c[0][1]);
      c[1][0] = wmma_acc(a1, b0, c[1][0]);
      c[1][1] = wmma_acc(a1, b1, c[1][1]);
    }
    __syncthreads();
  }
#pragma unroll
  for (int mi = 0; mi < 2; ++mi)
#pragma unroll
    for (int ni = 0; ni < 2; ++ni)
#pragma unroll
      for (int j = 0; j < 8; ++j) {
        int m = m0 + mh + mi * 16 + ((lane >> 4) << 3) + j;
        int n = n0 + nh + ni * 16 + (lane & 15);
        size_t idx = (size_t)m * N + n;
        if (EPI == 0) C[idx] = c[mi][ni][j];
        else          C[idx] += c[mi][ni][j];
      }
}

/* ---------------- gated FF GEMM: g = hx * silu(hg), bf16 out ----------------
   W1 is K x 6144; workgroup covers cols [n0,n0+64) and [n0+3072, n0+3072+64). */

__global__ __launch_bounds__(128) void k_gemm_gated(const float* __restrict__ A,
                                                    const float* __restrict__ W1,
                                                    u16* __restrict__ G,
                                                    int M, int K) {
  __shared__ u16 As[64 * 66];
  __shared__ u32 WLp[32 * 65];
  __shared__ u32 WRp[32 * 65];
  const int tid = threadIdx.x, lane = tid & 31, wv = tid >> 5;
  const int m0 = blockIdx.y * 64, n0 = blockIdx.x * 64;
  const int mh = (wv & 1) * 32, nh = (wv >> 1) * 32;
  v8f cL[2][2], cR[2][2];
#pragma unroll
  for (int a = 0; a < 2; ++a)
#pragma unroll
    for (int b2 = 0; b2 < 2; ++b2) { cL[a][b2] = zero8(); cR[a][b2] = zero8(); }

  const int r = tid >> 1, ch = (tid & 1) * 32;
  const int kp = tid >> 2, nb = (tid & 3) * 16;
  for (int k0 = 0; k0 < K; k0 += 64) {
    {
      const float* src = A + (size_t)(m0 + r) * K + k0 + ch;
      u32* da = (u32*)As + ((r * 66 + ch) >> 1);
#pragma unroll
      for (int i = 0; i < 8; ++i) {
        float4 v4 = ((const float4*)src)[i];
        da[i * 2] = pk2(v4.x, v4.y);
        da[i * 2 + 1] = pk2(v4.z, v4.w);
      }
      if (k0 + 64 < K) __builtin_prefetch((const char*)src + 256, 0, 1);
    }
    {
      const float* l0 = W1 + (size_t)(k0 + 2 * kp) * 6144 + n0 + nb;
      const float* l1 = l0 + 6144;
      const float* q0 = l0 + 3072;
      const float* q1 = l1 + 3072;
      u32* dl = WLp + kp * 65 + nb;
      u32* dr = WRp + kp * 65 + nb;
#pragma unroll
      for (int i = 0; i < 4; ++i) {
        float4 a4 = ((const float4*)l0)[i];
        float4 b4 = ((const float4*)l1)[i];
        float4 c4 = ((const float4*)q0)[i];
        float4 d4 = ((const float4*)q1)[i];
        dl[i * 4 + 0] = pk2(a4.x, b4.x); dl[i * 4 + 1] = pk2(a4.y, b4.y);
        dl[i * 4 + 2] = pk2(a4.z, b4.z); dl[i * 4 + 3] = pk2(a4.w, b4.w);
        dr[i * 4 + 0] = pk2(c4.x, d4.x); dr[i * 4 + 1] = pk2(c4.y, d4.y);
        dr[i * 4 + 2] = pk2(c4.z, d4.z); dr[i * 4 + 3] = pk2(c4.w, d4.w);
      }
      if (k0 + 64 < K) __builtin_prefetch((const char*)(l0 + (size_t)64 * 6144), 0, 1);
    }
    __syncthreads();
#pragma unroll
    for (int kk = 0; kk < 64; kk += 32) {
      v16bf a0 = ldsA(As, 66, mh, kk, lane);
      v16bf a1 = ldsA(As, 66, mh + 16, kk, lane);
      v16bf l0 = ldsBP(WLp, 65, nh, kk, lane);
      v16bf l1 = ldsBP(WLp, 65, nh + 16, kk, lane);
      v16bf r0 = ldsBP(WRp, 65, nh, kk, lane);
      v16bf r1 = ldsBP(WRp, 65, nh + 16, kk, lane);
      cL[0][0] = wmma_acc(a0, l0, cL[0][0]); cL[0][1] = wmma_acc(a0, l1, cL[0][1]);
      cL[1][0] = wmma_acc(a1, l0, cL[1][0]); cL[1][1] = wmma_acc(a1, l1, cL[1][1]);
      cR[0][0] = wmma_acc(a0, r0, cR[0][0]); cR[0][1] = wmma_acc(a0, r1, cR[0][1]);
      cR[1][0] = wmma_acc(a1, r0, cR[1][0]); cR[1][1] = wmma_acc(a1, r1, cR[1][1]);
    }
    __syncthreads();
  }
#pragma unroll
  for (int mi = 0; mi < 2; ++mi)
#pragma unroll
    for (int ni = 0; ni < 2; ++ni)
#pragma unroll
      for (int j = 0; j < 8; ++j) {
        int m = m0 + mh + mi * 16 + ((lane >> 4) << 3) + j;
        int n = n0 + nh + ni * 16 + (lane & 15);
        float hx = cL[mi][ni][j], hg = cR[mi][ni][j];
        float sg = 1.f / (1.f + __expf(-hg));
        G[(size_t)m * FFI + n] = f2bf(hx * hg * sg);
      }
}

/* ---------------- q: scale + rotary + pack bf16 [b,h][96][64] ---------------- */

__global__ __launch_bounds__(256) void k_qprep(const float* __restrict__ qt,
                                               u16* __restrict__ qh) {
  int p = blockIdx.x, b = blockIdx.y;    /* p 0..95 */
#pragma unroll
  for (int i = 0; i < 3; ++i) {
    int idx = threadIdx.x + i * 256;     /* 0..767 */
    int h = idx >> 6, d = idx & 63;
    u16 outv = 0;
    if (p < NTOK) {
      const float* row = qt + ((size_t)b * NTOK + p) * DIMM + h * 64;
      float v;
      if (d < 32) {
        int rr = d >> 1;
        float fr = __powf(10000.f, -(float)rr / 16.f);
        float ang = (float)p * fr;
        float cc = __cosf(ang), ss = __sinf(ang);
        float x0 = row[2 * rr] * 0.125f;       /* SCALE = 64^-0.5 */
        float x1 = row[2 * rr + 1] * 0.125f;
        v = (d & 1) ? (x1 * cc + x0 * ss) : (x0 * cc - x1 * ss);
      } else v = row[d] * 0.125f;
      outv = f2bf(v);
    }
    qh[(((size_t)b * HEADS + h) * NPAD + p) * 64 + d] = outv;
  }
}

/* ---------------- k/v: null-kv prepend + rotary(k) + pack bf16 ---------------- */

__global__ __launch_bounds__(64) void k_kvprep(const float* __restrict__ kvt,
                                               const float* __restrict__ nullkv,
                                               u16* __restrict__ kh,
                                               u16* __restrict__ vh) {
  int j = blockIdx.x, b = blockIdx.y;    /* j 0..95 */
  int d = threadIdx.x;
  float kk = 0.f, vv = 0.f;
  if (j == 0) { kk = nullkv[d]; vv = nullkv[64 + d]; }
  else if (j <= NTOK) {
    int p = j - 1;
    const float* row = kvt + ((size_t)b * NTOK + p) * 128;
    if (d < 32) {
      int rr = d >> 1;
      float fr = __powf(10000.f, -(float)rr / 16.f);
      float ang = (float)p * fr;
      float cc = __cosf(ang), ss = __sinf(ang);
      float x0 = row[2 * rr], x1 = row[2 * rr + 1];
      kk = (d & 1) ? (x1 * cc + x0 * ss) : (x0 * cc - x1 * ss);
    } else kk = row[d];
    vv = row[64 + d];
  }
  size_t o = ((size_t)b * NPAD + j) * 64 + d;
  kh[o] = f2bf(kk);
  vh[o] = f2bf(vv);
}

/* ---------------- pack V along K-pairs for WMMA B operand -------------------- */

__global__ __launch_bounds__(256) void k_vpack(const u16* __restrict__ vh,
                                               u32* __restrict__ vt) {
  int b = blockIdx.x;
  for (int i = threadIdx.x; i < 48 * 64; i += 256) {
    int kp = i >> 6, n = i & 63;
    u32 lo = vh[((size_t)b * NPAD + 2 * kp) * 64 + n];
    u32 hi = vh[((size_t)b * NPAD + 2 * kp + 1) * 64 + n];
    vt[(size_t)b * 48 * 64 + i] = lo | (hi << 16);
  }
}

/* ---------------- fused attention per (b,h): softmax(q k^T + bias) v ---------- */

__global__ __launch_bounds__(192) void k_attn(const u16* __restrict__ qh,
                                              const u16* __restrict__ kh,
                                              const u32* __restrict__ vt,
                                              const float* __restrict__ bias,
                                              const unsigned char* __restrict__ mask,
                                              u16* __restrict__ oh) {
  __shared__ u16 qs[96 * 66], ks[96 * 66];
  __shared__ u32 vsp[48 * 65];
  __shared__ u16 ps[96 * 98];
  const int bh = blockIdx.x, b = bh / HEADS, h = bh % HEADS;
  const int tid = threadIdx.x, lane = tid & 31, wv = tid >> 5;   /* wv: m-tile 0..5 */
  {
    const u32* qg = (const u32*)(qh + (size_t)bh * NPAD * 64);
    const u32* kg = (const u32*)(kh + (size_t)b * NPAD * 64);
    const u32* vg = vt + (size_t)b * 48 * 64;
    u32* q2 = (u32*)qs; u32* k2 = (u32*)ks;
#ifdef HAVE_ASYNC_LDS
    for (int i = tid; i < 96 * 32; i += 192) {
      int m = i >> 5, dp = i & 31;
      int dst = m * 33 + dp;
      async_cp_b32(&q2[dst], &qg[i]);
      async_cp_b32(&k2[dst], &kg[i]);
    }
    for (int i = tid; i < 48 * 64; i += 192) {
      int kp = i >> 6, n = i & 63;
      async_cp_b32(&vsp[kp * 65 + n], &vg[i]);
    }
    ASYNC_WAIT();
#else
    for (int i = tid; i < 96 * 32; i += 192) {
      int m = i >> 5, dp = i & 31;
      int dst = m * 33 + dp;
      q2[dst] = qg[i];
      k2[dst] = kg[i];
    }
    for (int i = tid; i < 48 * 64; i += 192) {
      int kp = i >> 6, n = i & 63;
      vsp[kp * 65 + n] = vg[i];
    }
#endif
  }
  __syncthreads();

  /* S = q @ k^T : 16x96 rows per wave */
  v8f s[6];
#pragma unroll
  for (int t = 0; t < 6; ++t) s[t] = zero8();
#pragma unroll
  for (int kk = 0; kk < 64; kk += 32) {
    v16bf a = ldsA(qs, 66, wv * 16, kk, lane);
#pragma unroll
    for (int t = 0; t < 6; ++t) {
      v16bf bb = ldsBT(ks, 66, t * 16, kk, lane);
      s[t] = wmma_acc(a, bb, s[t]);
    }
  }

  /* bias + mask + row softmax (rows spread over 16-lane halves) */
  const int mbase = wv * 16 + ((lane >> 4) << 3);
  const int ncol = lane & 15;
#pragma unroll
  for (int j = 0; j < 8; ++j) {
    int m = mbase + j;
    float mx = NEGF;
#pragma unroll
    for (int t = 0; t < 6; ++t) {
      int n = t * 16 + ncol;
      float v = s[t][j];
      if (m < NTOK && n < 82) v += bias[((size_t)h * NTOK + m) * 82 + n];
      bool km = (n == 0) || (n >= 78 && n < 82) ||
                (n >= 1 && n <= 77 && mask[(size_t)b * 77 + n - 1]);
      bool ok = (n <= m + 1) && (n < 82) && km;
      v = ok ? v : NEGF;
      s[t][j] = v;
      mx = fmaxf(mx, v);
    }
#pragma unroll
    for (int o = 1; o < 16; o <<= 1) mx = fmaxf(mx, __shfl_xor(mx, o, 32));
    float sum = 0.f;
#pragma unroll
    for (int t = 0; t < 6; ++t) {
      float e = __expf(s[t][j] - mx);
      s[t][j] = e;
      sum += e;
    }
#pragma unroll
    for (int o = 1; o < 16; o <<= 1) sum += __shfl_xor(sum, o, 32);
    float inv = 1.f / sum;
#pragma unroll
    for (int t = 0; t < 6; ++t) {
      int n = t * 16 + ncol;
      ps[m * 98 + n] = f2bf(s[t][j] * inv);
    }
  }
  __syncthreads();

  /* O = P @ V : 16x64 per wave, K = 96 */
  v8f o4[4];
#pragma unroll
  for (int t = 0; t < 4; ++t) o4[t] = zero8();
#pragma unroll
  for (int kk = 0; kk < 96; kk += 32) {
    v16bf a = ldsA(ps, 98, wv * 16, kk, lane);
#pragma unroll
    for (int t = 0; t < 4; ++t) {
      v16bf bb = ldsBP(vsp, 65, t * 16, kk, lane);
      o4[t] = wmma_acc(a, bb, o4[t]);
    }
  }
#pragma unroll
  for (int t = 0; t < 4; ++t)
#pragma unroll
    for (int j = 0; j < 8; ++j) {
      int m = mbase + j;
      if (m < NTOK) {
        int n = t * 16 + ncol;
        oh[((size_t)b * NTOK + m) * DIMM + h * 64 + n] = f2bf(o4[t][j]);
      }
    }
}

/* ---------------- driver ---------------- */

extern "C" void kernel_launch(void* const* d_in, const int* in_sizes, int n_in,
                              void* d_out, int out_size, void* d_ws, size_t ws_size,
                              hipStream_t stream) {
  (void)in_sizes; (void)n_in; (void)out_size; (void)ws_size;
  const float* rel_bias   = (const float*)d_in[0];
  const float* time_emb   = (const float*)d_in[1];
  const float* lq         = (const float*)d_in[2];
  const float* attn_g     = (const float*)d_in[3];
  const float* wq         = (const float*)d_in[4];
  const float* wkv        = (const float*)d_in[5];
  const float* null_kv    = (const float*)d_in[6];
  const float* wo         = (const float*)d_in[7];
  const float* attn_out_g = (const float*)d_in[8];
  const float* ff_g       = (const float*)d_in[9];
  const float* ff_w1      = (const float*)d_in[10];
  const float* ff_w2      = (const float*)d_in[11];
  const float* final_g    = (const float*)d_in[12];
  const float* proj       = (const float*)d_in[13];
  const float* img_emb    = (const float*)d_in[14];
  const float* txt_emb    = (const float*)d_in[15];
  const float* txt_enc    = (const float*)d_in[16];
  const int*   tsteps     = (const int*)d_in[17];
  const unsigned char* mask = (const unsigned char*)d_in[18];

  /* workspace layout */
  float* x    = (float*)d_ws;
  float* xn   = x + (size_t)ROWS * DIMM;
  float* tmp  = xn + (size_t)ROWS * DIMM;
  float* bias = tmp + (size_t)ROWS * DIMM;                 /* 12*81*82 */
  float* xe   = bias + 81920;
  u16* qh = (u16*)(xe + (size_t)BATCH * DIMM);
  u16* kh = qh + (size_t)BATCH * HEADS * NPAD * 64;
  u16* vh = kh + (size_t)BATCH * NPAD * 64;
  u16* oh = vh + (size_t)BATCH * NPAD * 64;
  u16* gb = oh + (size_t)ROWS * DIMM;
  u32* vt = (u32*)(gb + (size_t)ROWS * FFI);

  k_embed<<<dim3(NTOK, BATCH), 256, 0, stream>>>(txt_enc, txt_emb, time_emb, tsteps,
                                                 img_emb, lq, x);
  k_bias<<<dim3(NTOK), 96, 0, stream>>>(rel_bias, bias);

  for (int l = 0; l < 12; ++l) {
    const float* wq_l  = wq + (size_t)l * DIMM * DIMM;
    const float* wkv_l = wkv + (size_t)l * DIMM * 128;
    const float* wo_l  = wo + (size_t)l * DIMM * DIMM;
    const float* w1_l  = ff_w1 + (size_t)l * DIMM * 6144;
    const float* w2_l  = ff_w2 + (size_t)l * FFI * DIMM;
    const float* nk_l  = null_kv + (size_t)l * 128;

    k_ln<<<ROWS, 256, 0, stream>>>(x, DIMM, attn_g + l * DIMM, xn, DIMM, 0);
    k_gemm<0, 0><<<dim3(DIMM / 64, ROWS / 64), 128, 0, stream>>>(xn, wq_l, tmp,
                                                                 ROWS, DIMM, DIMM);
    k_qprep<<<dim3(NPAD, BATCH), 256, 0, stream>>>(tmp, qh);
    k_gemm<0, 0><<<dim3(128 / 64, ROWS / 64), 128, 0, stream>>>(xn, wkv_l, tmp,
                                                                ROWS, DIMM, 128);
    k_kvprep<<<dim3(NPAD, BATCH), 64, 0, stream>>>(tmp, nk_l, kh, vh);
    k_vpack<<<BATCH, 256, 0, stream>>>(vh, vt);
    k_attn<<<BATCH * HEADS, 192, 0, stream>>>(qh, kh, vt, bias, mask, oh);
    k_gemm<1, 0><<<dim3(DIMM / 64, ROWS / 64), 128, 0, stream>>>(oh, wo_l, tmp,
                                                                 ROWS, DIMM, DIMM);
    k_ln<<<ROWS, 256, 0, stream>>>(tmp, DIMM, attn_out_g + l * DIMM, x, DIMM, 1);
    k_ln<<<ROWS, 256, 0, stream>>>(x, DIMM, ff_g + l * DIMM, xn, DIMM, 0);
    k_gemm_gated<<<dim3(FFI / 64, ROWS / 64), 128, 0, stream>>>(xn, w1_l, gb,
                                                                ROWS, DIMM);
    k_gemm<1, 1><<<dim3(DIMM / 64, ROWS / 64), 128, 0, stream>>>(gb, w2_l, x,
                                                                 ROWS, FFI, DIMM);
  }

  /* final: LN on last token only, then 256x768x768 projection into d_out */
  k_ln<<<BATCH, 256, 0, stream>>>(x + (size_t)80 * DIMM, (long)NTOK * DIMM,
                                  final_g, xe, DIMM, 0);
  k_gemm<0, 0><<<dim3(DIMM / 64, BATCH / 64), 128, 0, stream>>>(xe, proj,
                                                                (float*)d_out,
                                                                BATCH, DIMM, DIMM);
}
